// FastSlowKimiDeltaAttention_46806553591959
// MI455X (gfx1250) — matrix operand order repaired and checked by reading, not compiled, not run on hardware
//
#include <hip/hip_runtime.h>
#include <hip/hip_bf16.h>
#include <math.h>

// Problem constants (match reference)
#define BSZ   2
#define TLEN  1024
#define HID_  2048
#define NH    16
#define DK_   128
#define DV_   128
#define HC_   2048            // H*DK = H*DV
#define MROWS (BSZ*TLEN)      // 2048 tokens

typedef __attribute__((ext_vector_type(16))) __bf16 v16bf;
typedef __attribute__((ext_vector_type(8)))  __bf16 v8bf;
typedef __attribute__((ext_vector_type(8)))  float  v8f;

__device__ __forceinline__ float sigmoidf_(float x) { return 1.0f / (1.0f + __expf(-x)); }
__device__ __forceinline__ float softplusf_(float x) { return (x > 20.0f) ? x : log1pf(__expf(x)); }

// ---------------------------------------------------------------------------
// fp32 -> bf16 cast (row-major copy)
__global__ void k_cast_bf16(const float* __restrict__ src, __hip_bfloat16* __restrict__ dst, int n) {
    int i = blockIdx.x * blockDim.x + threadIdx.x;
    if (i < n) dst[i] = __float2bfloat16(src[i]);
}

// fp32 [R,C] -> bf16 [C,R] transpose-cast (weights -> Bt layout for WMMA)
__global__ void k_transpose_bf16(const float* __restrict__ src, __hip_bfloat16* __restrict__ dst,
                                 int R, int C) {
    int i = blockIdx.x * blockDim.x + threadIdx.x;
    if (i < R * C) {
        int r = i / C, c = i % C;
        dst[(size_t)c * R + r] = __float2bfloat16(src[i]);
    }
}

// ---------------------------------------------------------------------------
// WMMA fragment loaders (wave32 layouts per CDNA5 ISA 7.12.2).
// A fragment, 16-bit A 16x32: lane half 0 -> K[+0..7] & K[+16..23],
//                              lane half 1 -> K[+8..15] & K[+24..31].
__device__ __forceinline__ v16bf load_frag_a(const __hip_bfloat16* arow_khalf) {
    v8bf lo = *reinterpret_cast<const v8bf*>(arow_khalf);
    v8bf hi = *reinterpret_cast<const v8bf*>(arow_khalf + 16);
    v16bf a;
#pragma unroll
    for (int i = 0; i < 8; ++i) { a[i] = lo[i]; a[i + 8] = hi[i]; }
    return a;
}
// B fragment, 16-bit B 32x16: lane half 0 -> K[+0..7] & K[+16..23],
// lane half 1 -> K[+8..15] & K[+24..31] for column n (Bt is [N,K] row-major).
__device__ __forceinline__ v16bf load_frag_b(const __hip_bfloat16* brow_khalf) {
    v8bf lo = *reinterpret_cast<const v8bf*>(brow_khalf);
    v8bf hi = *reinterpret_cast<const v8bf*>(brow_khalf + 16);
    v16bf b;
#pragma unroll
    for (int i = 0; i < 8; ++i) { b[i] = lo[i]; b[i + 8] = hi[i]; }
    return b;
}

// ---------------------------------------------------------------------------
// Simple bf16 GEMM: one 16x16 C tile per wave (used for small-N projections).
__global__ __launch_bounds__(32)
void k_gemm_bf16(const __hip_bfloat16* __restrict__ A,
                 const __hip_bfloat16* __restrict__ Bt,
                 float* __restrict__ C,
                 const float* __restrict__ bias,
                 int M, int N, int K) {
    const int lane = threadIdx.x;
    const int half = lane >> 4;
    const int l16  = lane & 15;
    const int tn = blockIdx.x, tm = blockIdx.y;
    const int m = tm * 16 + l16;
    const int n = tn * 16 + l16;
    const __hip_bfloat16* arow = A  + (size_t)m * K + half * 8;
    const __hip_bfloat16* brow = Bt + (size_t)n * K + half * 8;

    v8f acc;
#pragma unroll
    for (int i = 0; i < 8; ++i) acc[i] = 0.0f;

    for (int k0 = 0; k0 < K; k0 += 32) {
        v16bf a = load_frag_a(arow + k0);
        v16bf b = load_frag_b(brow + k0);
        if (k0 + 32 < K) {
            __builtin_prefetch((const void*)(arow + k0 + 32), 0, 1);
            __builtin_prefetch((const void*)(brow + k0 + 32), 0, 1);
        }
        acc = __builtin_amdgcn_wmma_f32_16x16x32_bf16(false, a, false, b,
                                                      (short)0, acc, false, false);
    }
    const float bv = bias ? bias[n] : 0.0f;
    const int mb = tm * 16 + half * 8;
#pragma unroll
    for (int r = 0; r < 8; ++r) C[(size_t)(mb + r) * N + n] = acc[r] + bv;
}

// ---------------------------------------------------------------------------
// Tiled bf16 GEMM: one wave computes a 32x64 C macro-tile (2 M-tiles x 4 N-tiles,
// 8 WMMAs per K-step) with double-buffered fragments so loads for K-step k0+32
// are in flight while the WMMAs for k0 execute. 1.5 b128 loads per WMMA.
// Requires M%32==0, N%64==0, K%32==0, K>=64.
__global__ __launch_bounds__(32)
void k_gemm_bf16_2x4(const __hip_bfloat16* __restrict__ A,
                     const __hip_bfloat16* __restrict__ Bt,
                     float* __restrict__ C,
                     const float* __restrict__ bias,
                     int M, int N, int K) {
    const int lane = threadIdx.x;
    const int half = lane >> 4;
    const int l16  = lane & 15;
    const int tn = blockIdx.x, tm = blockIdx.y;

    const __hip_bfloat16* arow[2];
#pragma unroll
    for (int i = 0; i < 2; ++i)
        arow[i] = A + (size_t)(tm * 32 + i * 16 + l16) * K + half * 8;
    const __hip_bfloat16* brow[4];
#pragma unroll
    for (int j = 0; j < 4; ++j)
        brow[j] = Bt + (size_t)(tn * 64 + j * 16 + l16) * K + half * 8;

    v8f acc[2][4];
#pragma unroll
    for (int i = 0; i < 2; ++i)
#pragma unroll
        for (int j = 0; j < 4; ++j)
#pragma unroll
            for (int r = 0; r < 8; ++r) acc[i][j][r] = 0.0f;

    // preload K-step 0
    v16bf a[2], b[4];
#pragma unroll
    for (int i = 0; i < 2; ++i) a[i] = load_frag_a(arow[i]);
#pragma unroll
    for (int j = 0; j < 4; ++j) b[j] = load_frag_b(brow[j]);

    for (int k0 = 0; k0 < K - 32; k0 += 32) {
        const int kn = k0 + 32;
        // issue next K-step's loads before consuming the current fragments
        v16bf na[2], nb[4];
#pragma unroll
        for (int i = 0; i < 2; ++i) na[i] = load_frag_a(arow[i] + kn);
#pragma unroll
        for (int j = 0; j < 4; ++j) nb[j] = load_frag_b(brow[j] + kn);
        if (kn + 32 < K) {   // gfx1250 global_prefetch_b8, two steps ahead
#pragma unroll
            for (int i = 0; i < 2; ++i)
                __builtin_prefetch((const void*)(arow[i] + kn + 32), 0, 1);
#pragma unroll
            for (int j = 0; j < 4; ++j)
                __builtin_prefetch((const void*)(brow[j] + kn + 32), 0, 1);
        }
#pragma unroll
        for (int i = 0; i < 2; ++i)
#pragma unroll
            for (int j = 0; j < 4; ++j)
                acc[i][j] = __builtin_amdgcn_wmma_f32_16x16x32_bf16(
                    false, a[i], false, b[j], (short)0, acc[i][j], false, false);
#pragma unroll
        for (int i = 0; i < 2; ++i) a[i] = na[i];
#pragma unroll
        for (int j = 0; j < 4; ++j) b[j] = nb[j];
    }
    // last K-step
#pragma unroll
    for (int i = 0; i < 2; ++i)
#pragma unroll
        for (int j = 0; j < 4; ++j)
            acc[i][j] = __builtin_amdgcn_wmma_f32_16x16x32_bf16(
                false, a[i], false, b[j], (short)0, acc[i][j], false, false);

    // epilogue: C layout lane=N column, VGPR r -> row (half*8 + r) of the 16x16 tile
#pragma unroll
    for (int i = 0; i < 2; ++i) {
        const int mb = tm * 32 + i * 16 + half * 8;
#pragma unroll
        for (int j = 0; j < 4; ++j) {
            const int n = tn * 64 + j * 16 + l16;
            const float bv = bias ? bias[n] : 0.0f;
#pragma unroll
            for (int r = 0; r < 8; ++r)
                C[(size_t)(mb + r) * N + n] = acc[i][j][r] + bv;
        }
    }
}

// ---------------------------------------------------------------------------
// causal depthwise conv (K=4) + SiLU over [B,T,HC]
__global__ void k_conv_silu(const float* __restrict__ y, const float* __restrict__ w,
                            float* __restrict__ out, int total) {
    int i = blockIdx.x * blockDim.x + threadIdx.x;
    if (i >= total) return;
    int c  = i & (HC_ - 1);
    int bt = i >> 11;               // HC_ == 2^11
    int t  = bt & (TLEN - 1);
    float acc = 0.0f;
#pragma unroll
    for (int j = 0; j < 4; ++j) {
        int tt = t - 3 + j;
        if (tt >= 0) acc += y[(size_t)(bt - 3 + j) * HC_ + c] * w[c * 4 + j];
    }
    out[i] = acc * sigmoidf_(acc);
}

// gf/gs from g_base,g_delta (in place: gb<-gf, gd<-gs)
__global__ void k_gates(float* __restrict__ gb, float* __restrict__ gd,
                        const float* __restrict__ A_log, const float* __restrict__ dtb,
                        int total) {
    int i = blockIdx.x * blockDim.x + threadIdx.x;
    if (i >= total) return;
    int c = i & (HC_ - 1);
    float a = -__expf(A_log[c >> 7]);            // head = c / 128
    float x1 = gb[i], x2 = gd[i], d = dtb[c];
    gb[i] = a * softplusf_(x1 + x2 + d);
    gd[i] = a * softplusf_(x1 - x2 + d);
}

// beta_fast/slow + lambda (in place)
__global__ void k_beta_lam(float* __restrict__ bb, float* __restrict__ bd,
                           float* __restrict__ lam, int total) {
    int i = blockIdx.x * blockDim.x + threadIdx.x;
    if (i >= total) return;
    float a = bb[i], d = bd[i];
    bb[i] = sigmoidf_(a + d);
    bd[i] = sigmoidf_(a - d);
    lam[i] = sigmoidf_(lam[i]);
}

// L2-normalize contiguous groups of 128 (q/k heads), in place. grid = M*H, block = 128
__global__ __launch_bounds__(128) void k_l2norm(float* __restrict__ x) {
    __shared__ float red[128];
    size_t base = (size_t)blockIdx.x * 128;
    int e = threadIdx.x;
    float v = x[base + e];
    red[e] = v * v;
    __syncthreads();
    for (int s = 64; s > 0; s >>= 1) {
        if (e < s) red[e] += red[e + s];
        __syncthreads();
    }
    x[base + e] = v * rsqrtf(red[0] + 1e-6f);
}

// ---------------------------------------------------------------------------
// dual-state gated delta-rule recurrence.
// grid = B*H*2 blocks; 128 threads; thread e owns state column S[:,e] in registers.
__global__ __launch_bounds__(128)
void k_fkda(const float* __restrict__ Q, const float* __restrict__ Kk, const float* __restrict__ V,
            const float* __restrict__ GF, const float* __restrict__ GS,
            const float* __restrict__ BF, const float* __restrict__ BS,
            float* __restrict__ OF, float* __restrict__ OS) {
    int idx = blockIdx.x;               // b*H*2 + h*2 + s
    int s = idx & 1;
    int h = (idx >> 1) & (NH - 1);
    int b = idx >> 5;
    const float* G  = s ? GS : GF;
    const float* Bt = s ? BS : BF;
    float*       O  = s ? OS : OF;
    const int e = threadIdx.x;

    __shared__ float ks_[128], qs_[128], eg_[128];
    float S[128];
#pragma unroll
    for (int d = 0; d < 128; ++d) S[d] = 0.0f;

    const float scale = 0.08838834764831845f;   // 128^-0.5

    for (int t = 0; t < TLEN; ++t) {
        size_t row  = (size_t)b * TLEN + t;
        size_t base = row * (size_t)HC_ + (size_t)h * 128;
        ks_[e] = Kk[base + e];
        qs_[e] = Q[base + e] * scale;
        eg_[e] = __expf(G[base + e]);
        __syncthreads();

        float vt  = V[base + e];
        float bet = Bt[row * NH + h];
        float mem = 0.0f;
#pragma unroll
        for (int d = 0; d < 128; ++d) {
            float sd = S[d] * eg_[d];
            mem = fmaf(ks_[d], sd, mem);
            S[d] = sd;
        }
        float u = bet * (vt - mem);
        float o = 0.0f;
#pragma unroll
        for (int d = 0; d < 128; ++d) {
            float sd = fmaf(ks_[d], u, S[d]);
            o = fmaf(qs_[d], sd, o);
            S[d] = sd;
        }
        O[base + e] = o;
        __syncthreads();
    }
}

// mix fast/slow, RMS norm over DV, * onorm_w, * sigmoid(gate). grid = M*H, block = 128
__global__ __launch_bounds__(128)
void k_mix_norm_gate(const float* __restrict__ of, const float* __restrict__ os,
                     const float* __restrict__ lam, const float* __restrict__ gate,
                     const float* __restrict__ onw, float* __restrict__ out) {
    __shared__ float red[128];
    size_t base = (size_t)blockIdx.x * 128;
    int bh = blockIdx.x;
    int row = bh >> 4, h = bh & (NH - 1);
    float l = lam[(size_t)row * NH + h];
    int e = threadIdx.x;
    float o = l * of[base + e] + (1.0f - l) * os[base + e];
    red[e] = o * o;
    __syncthreads();
    for (int s = 64; s > 0; s >>= 1) {
        if (e < s) red[e] += red[e + s];
        __syncthreads();
    }
    float ms = red[0] * (1.0f / 128.0f);
    out[base + e] = o * rsqrtf(ms + 1e-5f) * onw[e] * sigmoidf_(gate[base + e]);
}

// ---------------------------------------------------------------------------
extern "C" void kernel_launch(void* const* d_in, const int* in_sizes, int n_in,
                              void* d_out, int out_size, void* d_ws, size_t ws_size,
                              hipStream_t stream) {
    (void)in_sizes; (void)n_in; (void)out_size; (void)ws_size;
    const float* x       = (const float*)d_in[0];
    const float* Wq      = (const float*)d_in[1];
    const float* Wk      = (const float*)d_in[2];
    const float* Wv      = (const float*)d_in[3];
    const float* conv_wq = (const float*)d_in[4];
    const float* conv_wk = (const float*)d_in[5];
    const float* conv_wv = (const float*)d_in[6];
    const float* Wgb1    = (const float*)d_in[7];
    const float* Wgb2    = (const float*)d_in[8];
    const float* Wgd1    = (const float*)d_in[9];
    const float* Wgd2    = (const float*)d_in[10];
    const float* Wbb     = (const float*)d_in[11];
    const float* Wbd     = (const float*)d_in[12];
    const float* Wlam    = (const float*)d_in[13];
    const float* A_log   = (const float*)d_in[14];
    const float* dt_bias = (const float*)d_in[15];
    const float* Wg1     = (const float*)d_in[16];
    const float* Wg2     = (const float*)d_in[17];
    const float* bg2     = (const float*)d_in[18];
    const float* onorm_w = (const float*)d_in[19];
    const float* Wo      = (const float*)d_in[20];
    float* out = (float*)d_out;

    // bump allocator over d_ws
    char* ws = (char*)d_ws;
    size_t off = 0;
    auto alloc = [&](size_t bytes) -> char* {
        char* p = ws + off;
        off += (bytes + 255) & ~(size_t)255;
        return p;
    };
    const size_t MH = (size_t)MROWS * HC_;            // 2048*2048
    __hip_bfloat16* XB = (__hip_bfloat16*)alloc(MH * 2);       // x in bf16 (persist)
    __hip_bfloat16* TW = (__hip_bfloat16*)alloc(MH * 2);       // transposed weight (reused)
    __hip_bfloat16* CB = (__hip_bfloat16*)alloc(MH * 2);       // cast buffer (reused)
    float* PRE  = (float*)alloc(MH * 4);                        // pre-conv projection (reused)
    float* QC   = (float*)alloc(MH * 4);
    float* KC   = (float*)alloc(MH * 4);
    float* VC   = (float*)alloc(MH * 4);
    float* GFb  = (float*)alloc(MH * 4);                        // g_base -> gf
    float* GSb  = (float*)alloc(MH * 4);                        // g_delta -> gs
    float* GATE = (float*)alloc(MH * 4);
    float* OFb  = (float*)alloc(MH * 4);
    float* OSb  = (float*)alloc(MH * 4);
    float* OM   = (float*)alloc(MH * 4);
    float* TMP1 = (float*)alloc((size_t)MROWS * 128 * 4);
    float* BBb  = (float*)alloc((size_t)MROWS * NH * 4);
    float* BDb  = (float*)alloc((size_t)MROWS * NH * 4);
    float* LAMR = (float*)alloc((size_t)MROWS * NH * 4);

    const int TPB = 256;
    auto grid1 = [&](size_t n) { return dim3((unsigned)((n + TPB - 1) / TPB)); };
    auto gemm = [&](const __hip_bfloat16* A, const __hip_bfloat16* Btw, float* Cc,
                    const float* bias, int M, int N, int K) {
        if ((M % 32 == 0) && (N % 64 == 0) && (K >= 64)) {
            k_gemm_bf16_2x4<<<dim3(N / 64, M / 32), dim3(32), 0, stream>>>(A, Btw, Cc, bias, M, N, K);
        } else {
            k_gemm_bf16<<<dim3(N / 16, M / 16), dim3(32), 0, stream>>>(A, Btw, Cc, bias, M, N, K);
        }
    };
    auto tW = [&](const float* W, int R, int C) {
        k_transpose_bf16<<<grid1((size_t)R * C), TPB, 0, stream>>>(W, TW, R, C);
    };

    // x -> bf16
    k_cast_bf16<<<grid1(MH), TPB, 0, stream>>>(x, XB, (int)MH);

    // q/k/v projections + conv + SiLU
    tW(Wq, HID_, HC_); gemm(XB, TW, PRE, nullptr, MROWS, HC_, HID_);
    k_conv_silu<<<grid1(MH), TPB, 0, stream>>>(PRE, conv_wq, QC, (int)MH);
    tW(Wk, HID_, HC_); gemm(XB, TW, PRE, nullptr, MROWS, HC_, HID_);
    k_conv_silu<<<grid1(MH), TPB, 0, stream>>>(PRE, conv_wk, KC, (int)MH);
    tW(Wv, HID_, HC_); gemm(XB, TW, PRE, nullptr, MROWS, HC_, HID_);
    k_conv_silu<<<grid1(MH), TPB, 0, stream>>>(PRE, conv_wv, VC, (int)MH);

    // low-rank decay gates: g_base, g_delta
    tW(Wgb1, HID_, 128); gemm(XB, TW, TMP1, nullptr, MROWS, 128, HID_);
    k_cast_bf16<<<grid1((size_t)MROWS * 128), TPB, 0, stream>>>(TMP1, CB, MROWS * 128);
    tW(Wgb2, 128, HC_);  gemm(CB, TW, GFb, nullptr, MROWS, HC_, 128);
    tW(Wgd1, HID_, 128); gemm(XB, TW, TMP1, nullptr, MROWS, 128, HID_);
    k_cast_bf16<<<grid1((size_t)MROWS * 128), TPB, 0, stream>>>(TMP1, CB, MROWS * 128);
    tW(Wgd2, 128, HC_);  gemm(CB, TW, GSb, nullptr, MROWS, HC_, 128);
    k_gates<<<grid1(MH), TPB, 0, stream>>>(GFb, GSb, A_log, dt_bias, (int)MH);

    // beta fast/slow and lambda (N=16 path -> simple kernel)
    tW(Wbb,  HID_, NH); gemm(XB, TW, BBb,  nullptr, MROWS, NH, HID_);
    tW(Wbd,  HID_, NH); gemm(XB, TW, BDb,  nullptr, MROWS, NH, HID_);
    tW(Wlam, HID_, NH); gemm(XB, TW, LAMR, nullptr, MROWS, NH, HID_);
    k_beta_lam<<<grid1((size_t)MROWS * NH), TPB, 0, stream>>>(BBb, BDb, LAMR, MROWS * NH);

    // output gate: (x@Wg1)@Wg2 + bg2
    tW(Wg1, HID_, 128); gemm(XB, TW, TMP1, nullptr, MROWS, 128, HID_);
    k_cast_bf16<<<grid1((size_t)MROWS * 128), TPB, 0, stream>>>(TMP1, CB, MROWS * 128);
    tW(Wg2, 128, HC_);  gemm(CB, TW, GATE, bg2, MROWS, HC_, 128);

    // q/k L2 norm (per head of 128)
    k_l2norm<<<dim3(MROWS * NH), dim3(128), 0, stream>>>(QC);
    k_l2norm<<<dim3(MROWS * NH), dim3(128), 0, stream>>>(KC);

    // dual-state gated delta-rule recurrence
    k_fkda<<<dim3(BSZ * NH * 2), dim3(128), 0, stream>>>(QC, KC, VC, GFb, GSb, BBb, BDb, OFb, OSb);

    // mix + gated RMS norm
    k_mix_norm_gate<<<dim3(MROWS * NH), dim3(128), 0, stream>>>(OFb, OSb, LAMR, GATE, onorm_w, OM);

    // output projection
    k_cast_bf16<<<grid1(MH), TPB, 0, stream>>>(OM, CB, (int)MH);
    tW(Wo, HC_, HID_);
    gemm(CB, TW, out, nullptr, MROWS, HID_, HC_);
}
